// FeedForward_6734508720308
// MI455X (gfx1250) — 8-bit matrix operand order repaired and checked by reading, compiled, not run on hardware
//
#include <hip/hip_runtime.h>
#include <hip/hip_bf16.h>
#include <stdint.h>


typedef __attribute__((ext_vector_type(8))) int v8i;

#define DEV __device__ __forceinline__

constexpr int Bk = 4, Sk = 2048, Hk = 2048, Fk = 5632;
constexpr int Mtok = Bk * Sk;
constexpr long long NW1 = 2LL * Fk * Hk;
constexpr long long NW2 = (long long)Hk * Fk;

DEV v8i wmma_iu8(v8i a, v8i b, v8i c) {
  return __builtin_amdgcn_wmma_i32_16x16x64_iu8(true, a, true, b, c, false, false);
}

DEV v8i load_a_frag(const signed char* __restrict__ A, int lda, int m0, int k0, int lane) {
  const int row  = m0 + (lane & 15);
  const int koff = (lane >> 4) * 16;
  const signed char* p = A + (size_t)row * (size_t)lda + k0 + koff;
  v8i f = {};
#pragma unroll
  for (int j = 0; j < 4; ++j) {
    uint2 d = *(const uint2*)(p + 32 * (j >> 1) + 8 * (j & 1));
    f[2 * j]     = (int)d.x;
    f[2 * j + 1] = (int)d.y;
  }
  return f;
}

DEV v8i load_b_frag(const signed char* __restrict__ W, int ldb, int n0, int k0, int lane) {
  const int row  = n0 + (lane & 15);
  const int koff = (lane >> 4) * 16;
  const signed char* p = W + (size_t)row * (size_t)ldb + k0 + koff;
  int4 lo = *(const int4*)(p);
  int4 hi = *(const int4*)(p + 32);
  v8i f = {};
  f[0] = lo.x; f[1] = lo.y; f[2] = lo.z; f[3] = lo.w;
  f[4] = hi.x; f[5] = hi.y; f[6] = hi.z; f[7] = hi.w;
  return f;
}

DEV float block_sum(float v, float* sm) {
  const int t = threadIdx.x;
  sm[t] = v;
  __syncthreads();
#pragma unroll
  for (int s = 128; s > 0; s >>= 1) {
    if (t < s) sm[t] += sm[t + s];
    __syncthreads();
  }
  float r = sm[0];
  __syncthreads();
  return r;
}

DEV float block_max(float v, float* sm) {
  const int t = threadIdx.x;
  sm[t] = v;
  __syncthreads();
#pragma unroll
  for (int s = 128; s > 0; s >>= 1) {
    if (t < s) sm[t] = fmaxf(sm[t], sm[t + s]);
    __syncthreads();
  }
  float r = sm[0];
  __syncthreads();
  return r;
}

__global__ void __launch_bounds__(256) wabs_partial(
    const float* __restrict__ w1, long long n1,
    const float* __restrict__ w2, long long n2,
    float* __restrict__ partials) {
  __shared__ float sm[256];
  const int b = blockIdx.x;
  const float* w;
  long long n, bi, nb;
  if (b < 512) { w = w1; n = n1; bi = b;       nb = 512; }
  else         { w = w2; n = n2; bi = b - 512; nb = 256; }
  float s = 0.f;
  for (long long i = bi * 256 + threadIdx.x; i < n; i += nb * 256)
    s += fabsf(w[i]);
  s = block_sum(s, sm);
  if (threadIdx.x == 0) partials[b] = s;
}

__global__ void wabs_final(float* __restrict__ hdr) {
  __shared__ float sm[256];
  float* partials = hdr + 16;
  const int t = threadIdx.x;
  float s1 = partials[t] + partials[t + 256];
  s1 = block_sum(s1, sm);
  float s2 = partials[512 + t];
  s2 = block_sum(s2, sm);
  if (t == 0) {
    float isw1 = fmaxf(s1 / (float)NW1, 1e-5f);
    float isw2 = fmaxf(s2 / (float)NW2, 1e-5f);
    hdr[0] = 1.f / isw1; hdr[1] = isw1;
    hdr[2] = 1.f / isw2; hdr[3] = isw2;
  }
}

DEV signed char ternq(float v, float sw) {
  float r = rintf(v * sw);
  r = fminf(1.f, fmaxf(-1.f, r));
  return (signed char)(int)r;
}

__global__ void __launch_bounds__(256) wquant(
    const float* __restrict__ w, signed char* __restrict__ q,
    long long nquads, const float* __restrict__ hdr, int which) {
  const float sw = hdr[which];
  const float4* w4 = (const float4*)w;
  char4* q4 = (char4*)q;
  long long i      = (long long)blockIdx.x * blockDim.x + threadIdx.x;
  long long stride = (long long)gridDim.x * blockDim.x;
  for (; i < nquads; i += stride) {
    float4 v = w4[i];
    char4 o;
    o.x = ternq(v.x, sw);
    o.y = ternq(v.y, sw);
    o.z = ternq(v.z, sw);
    o.w = ternq(v.w, sw);
    q4[i] = o;
  }
}

__global__ void __launch_bounds__(256) rmsnorm_quant(
    const float* __restrict__ x, const float* __restrict__ nw,
    signed char* __restrict__ xq, float* __restrict__ isx) {
  __shared__ float sm[256];
  const int m = blockIdx.x;
  const int t = threadIdx.x;
  const float* row = x + (size_t)m * Hk;
  float v[8];
  float ss = 0.f;
#pragma unroll
  for (int k = 0; k < 8; ++k) { v[k] = row[t + 256 * k]; ss += v[k] * v[k]; }
  ss = block_sum(ss, sm);
  const float rinv = rsqrtf(ss / (float)Hk + 1e-5f);
  float mx = 0.f;
#pragma unroll
  for (int k = 0; k < 8; ++k) {
    v[k] *= rinv * nw[t + 256 * k];
    mx = fmaxf(mx, fabsf(v[k]));
  }
  mx = block_max(mx, sm);
  const float am = fmaxf(mx, 1e-5f);
  const float s = 127.f / am;
#pragma unroll
  for (int k = 0; k < 8; ++k) {
    float r = rintf(v[k] * s);
    r = fminf(127.f, fmaxf(-128.f, r));
    xq[(size_t)m * Hk + t + 256 * k] = (signed char)(int)r;
  }
  if (t == 0) isx[m] = 1.f / s;
}

__global__ void __launch_bounds__(256) bitnet_gemm1(
    const signed char* __restrict__ xq, const float* __restrict__ isx,
    const signed char* __restrict__ wq, const float* __restrict__ hdr,
    float* __restrict__ hidden) {
  const int lane = threadIdx.x & 31;
  const int wave = threadIdx.x >> 5;
  const int wm = wave >> 1;
  const int wn = wave & 1;
  const int m0 = blockIdx.y * 128 + wm * 32;
  const int f0 = blockIdx.x * 64 + wn * 32;
  const float isw = hdr[1];

  v8i ag[2][2], au[2][2];
#pragma unroll
  for (int i = 0; i < 2; ++i)
#pragma unroll
    for (int j = 0; j < 2; ++j) { ag[i][j] = (v8i){}; au[i][j] = (v8i){}; }

  for (int k0 = 0; k0 < Hk; k0 += 64) {
    v8i a0  = load_a_frag(xq, Hk, m0,      k0, lane);
    v8i a1  = load_a_frag(xq, Hk, m0 + 16, k0, lane);
    v8i bg0 = load_b_frag(wq, Hk, f0,           k0, lane);
    v8i bg1 = load_b_frag(wq, Hk, f0 + 16,      k0, lane);
    v8i bu0 = load_b_frag(wq, Hk, f0 + Fk,      k0, lane);
    v8i bu1 = load_b_frag(wq, Hk, f0 + Fk + 16, k0, lane);
    ag[0][0] = wmma_iu8(a0, bg0, ag[0][0]);
    ag[0][1] = wmma_iu8(a0, bg1, ag[0][1]);
    ag[1][0] = wmma_iu8(a1, bg0, ag[1][0]);
    ag[1][1] = wmma_iu8(a1, bg1, ag[1][1]);
    au[0][0] = wmma_iu8(a0, bu0, au[0][0]);
    au[0][1] = wmma_iu8(a0, bu1, au[0][1]);
    au[1][0] = wmma_iu8(a1, bu0, au[1][0]);
    au[1][1] = wmma_iu8(a1, bu1, au[1][1]);
  }

  const int rbase = (lane >> 4) * 8;
  const int n = lane & 15;
#pragma unroll
  for (int i = 0; i < 2; ++i) {
#pragma unroll
    for (int r = 0; r < 8; ++r) {
      const int m = m0 + i * 16 + rbase + r;
      const float sc = isx[m] * isw;
#pragma unroll
      for (int j = 0; j < 2; ++j) {
        const int f = f0 + j * 16 + n;
        float g = (float)ag[i][j][r] * sc;
        float u = (float)au[i][j][r] * sc;
        float rg = g > 0.f ? g : 0.f;
        hidden[(size_t)m * Fk + f] = rg * rg * u;
      }
    }
  }
}

__global__ void __launch_bounds__(256) hidden_quant(
    const float* __restrict__ hid, signed char* __restrict__ hq,
    float* __restrict__ ish) {
  __shared__ float sm[256];
  const int m = blockIdx.x;
  const int t = threadIdx.x;
  const float* row = hid + (size_t)m * Fk;
  float v[22];
  float mx = 0.f;
#pragma unroll
  for (int k = 0; k < 22; ++k) { v[k] = row[t + 256 * k]; mx = fmaxf(mx, fabsf(v[k])); }
  mx = block_max(mx, sm);
  const float am = fmaxf(mx, 1e-5f);
  const float s = 127.f / am;
#pragma unroll
  for (int k = 0; k < 22; ++k) {
    float r = rintf(v[k] * s);
    r = fminf(127.f, fmaxf(-128.f, r));
    hq[(size_t)m * Fk + t + 256 * k] = (signed char)(int)r;
  }
  if (t == 0) ish[m] = 1.f / s;
}

__global__ void __launch_bounds__(256) bitnet_gemm2(
    const signed char* __restrict__ hq, const float* __restrict__ ish,
    const signed char* __restrict__ wq, const float* __restrict__ hdr,
    float* __restrict__ out) {
  const int lane = threadIdx.x & 31;
  const int wave = threadIdx.x >> 5;
  const int wm = wave >> 1;
  const int wn = wave & 1;
  const int m0 = blockIdx.y * 128 + wm * 32;
  const int n0 = blockIdx.x * 64 + wn * 32;
  const float isw = hdr[3];

  v8i acc[2][2];
#pragma unroll
  for (int i = 0; i < 2; ++i)
#pragma unroll
    for (int j = 0; j < 2; ++j) acc[i][j] = (v8i){};

  for (int k0 = 0; k0 < Fk; k0 += 64) {
    v8i a0 = load_a_frag(hq, Fk, m0,      k0, lane);
    v8i a1 = load_a_frag(hq, Fk, m0 + 16, k0, lane);
    v8i b0 = load_b_frag(wq, Fk, n0,      k0, lane);
    v8i b1 = load_b_frag(wq, Fk, n0 + 16, k0, lane);
    acc[0][0] = wmma_iu8(a0, b0, acc[0][0]);
    acc[0][1] = wmma_iu8(a0, b1, acc[0][1]);
    acc[1][0] = wmma_iu8(a1, b0, acc[1][0]);
    acc[1][1] = wmma_iu8(a1, b1, acc[1][1]);
  }

  const int rbase = (lane >> 4) * 8;
  const int n = lane & 15;
#pragma unroll
  for (int i = 0; i < 2; ++i) {
#pragma unroll
    for (int r = 0; r < 8; ++r) {
      const int m = m0 + i * 16 + rbase + r;
      const float sc = ish[m] * isw;
#pragma unroll
      for (int j = 0; j < 2; ++j) {
        const int nn = n0 + j * 16 + n;
        out[(size_t)m * Hk + nn] = (float)acc[i][j][r] * sc;
      }
    }
  }
}

extern "C" void kernel_launch(void* const* d_in, const int* in_sizes, int n_in,
                              void* d_out, int out_size, void* d_ws, size_t ws_size,
                              hipStream_t stream) {
  const float* x  = (const float*)d_in[0];
  const float* w1 = (const float*)d_in[1];
  const float* w2 = (const float*)d_in[2];
  const float* nw = (const float*)d_in[3];
  float* out = (float*)d_out;

  char* ws = (char*)d_ws;
  size_t off = 0;
  auto carve = [&](size_t bytes) -> void* {
    void* p = ws + off;
    off = (off + bytes + 255) & ~(size_t)255;
    return p;
  };

  float*        hdr    = (float*)       carve(4096);
  signed char*  xq     = (signed char*) carve((size_t)Mtok * Hk);
  float*        isx    = (float*)       carve((size_t)Mtok * 4);
  signed char*  wq1    = (signed char*) carve((size_t)NW1);
  signed char*  wq2    = (signed char*) carve((size_t)NW2);
  signed char*  hq     = (signed char*) carve((size_t)Mtok * Fk);
  float*        ish    = (float*)       carve((size_t)Mtok * 4);
  float*        hidden = (float*)       carve((size_t)Mtok * Fk * 4);
  (void)ws_size;

  wabs_partial<<<768, 256, 0, stream>>>(w1, NW1, w2, NW2, hdr + 16);
  wabs_final<<<1, 256, 0, stream>>>(hdr);

  wquant<<<2048, 256, 0, stream>>>(w1, wq1, NW1 / 4, hdr, 0);
  wquant<<<1024, 256, 0, stream>>>(w2, wq2, NW2 / 4, hdr, 2);

  rmsnorm_quant<<<Mtok, 256, 0, stream>>>(x, nw, xq, isx);

  bitnet_gemm1<<<dim3(Fk / 64, Mtok / 128), 256, 0, stream>>>(xq, isx, wq1, hdr, hidden);

  hidden_quant<<<Mtok, 256, 0, stream>>>(hidden, hq, ish);

  bitnet_gemm2<<<dim3(Hk / 64, Mtok / 128), 256, 0, stream>>>(hq, ish, wq2, hdr, out);
}
